// GPTModel_26353919328823
// MI455X (gfx1250) — matrix operand order repaired and checked by reading, not compiled, not run on hardware
//
#include <hip/hip_runtime.h>
#include <hip/hip_bf16.h>
#include <math.h>

// ---------------------------------------------------------------------------
// CDNA5 (gfx1250) GPT-2 forward. wave32, WMMA f32_16x16x32_f16.
// Activations kept f16 end-to-end (producers convert in epilogue); weights
// fp32 in HBM, converted to f16 while staged to LDS (GEMMs are compute-bound
// at ~1024 FLOP/byte so the cvt VALU co-executes under WMMA).
// ---------------------------------------------------------------------------

typedef __attribute__((ext_vector_type(16))) _Float16 v16h;
typedef __attribute__((ext_vector_type(8)))  _Float16 v8h;
typedef __attribute__((ext_vector_type(8)))  float    v8f;
typedef __attribute__((ext_vector_type(4)))  int      v4i;

#if defined(__AMDGCN__) && __has_builtin(__builtin_amdgcn_global_load_async_to_lds_b128)
#define USE_ASYNC_COPY 1
#else
#define USE_ASYNC_COPY 0
#endif

#if USE_ASYNC_COPY
static __device__ __forceinline__ void async_copy_b128(const void* g, void* l) {
  // Probe-derived prototype: (v4i* global_src, v4i* lds_dst, imm offset, imm cpol)
  __builtin_amdgcn_global_load_async_to_lds_b128(
      (v4i*)(uintptr_t)g, (v4i*)(uintptr_t)l, 0, 0);
}
static __device__ __forceinline__ void wait_async0() {
#if __has_builtin(__builtin_amdgcn_s_wait_asynccnt)
  __builtin_amdgcn_s_wait_asynccnt(0);
#else
  asm volatile("s_wait_asynccnt 0x0" ::: "memory");
#endif
}
#endif

static __device__ __forceinline__ v8f wmma_f16(v16h a, v16h b, v8f c) {
  // (neg_a, A, neg_b, B, c_mod, C, reuse_a, reuse_b)
  return __builtin_amdgcn_wmma_f32_16x16x32_f16(false, a, false, b,
                                                (short)0, c, false, false);
}

// ---------------------------------------------------------------------------
// Embedding: x[b,s,:] = tok_emb[tokens[b,s],:] + pos_emb[s,:]   (f32 residual)
// ---------------------------------------------------------------------------
__global__ void embed_k(const int* __restrict__ tok,
                        const float* __restrict__ te,
                        const float* __restrict__ pe,
                        float* __restrict__ x, int total, int D, int S) {
  for (int i = blockIdx.x * blockDim.x + threadIdx.x; i < total;
       i += gridDim.x * blockDim.x) {
    int bs = i / D;
    int d  = i - bs * D;
    int s  = bs % S;
    x[i] = te[(size_t)tok[bs] * D + d] + pe[(size_t)s * D + d];
  }
}

// ---------------------------------------------------------------------------
// LayerNorm: f32 in, f16 out (always feeds a WMMA GEMM A operand).
// ---------------------------------------------------------------------------
__global__ __launch_bounds__(256) void layernorm_k(const float* __restrict__ x,
                                                   const float* __restrict__ g,
                                                   const float* __restrict__ b,
                                                   _Float16* __restrict__ y,
                                                   int D) {
  __shared__ float red0[8], red1[8];
  const int row = blockIdx.x;
  const float* xr = x + (size_t)row * D;
  float s = 0.f, s2 = 0.f;
  for (int d = threadIdx.x; d < D; d += blockDim.x) {
    float v = xr[d];
    s += v; s2 += v * v;
  }
  for (int m = 16; m >= 1; m >>= 1) {
    s  += __shfl_xor(s,  m, 32);
    s2 += __shfl_xor(s2, m, 32);
  }
  const int wave = threadIdx.x >> 5, lane = threadIdx.x & 31;
  if (lane == 0) { red0[wave] = s; red1[wave] = s2; }
  __syncthreads();
  if (wave == 0) {
    s  = (lane < 8) ? red0[lane] : 0.f;
    s2 = (lane < 8) ? red1[lane] : 0.f;
    for (int m = 4; m >= 1; m >>= 1) {
      s  += __shfl_xor(s,  m, 8);
      s2 += __shfl_xor(s2, m, 8);
    }
    if (lane == 0) { red0[0] = s; red1[0] = s2; }
  }
  __syncthreads();
  const float mean = red0[0] / (float)D;
  const float var  = red1[0] / (float)D - mean * mean;
  const float rstd = rsqrtf(var + 1e-5f);
  for (int d = threadIdx.x; d < D; d += blockDim.x)
    y[(size_t)row * D + d] = (_Float16)((xr[d] - mean) * rstd * g[d] + b[d]);
}

// ---------------------------------------------------------------------------
// GEMM: out = act( A[M,K](f16) @ W[K,N](f32) + bias ) (+ residual, f32)
// Output either f16 (Ch, activation consumer) or f32 (Cf, residual/logits).
// WG tile 128x128, 8 waves (4M x 2N), per-wave 32x64 (2x4 WMMA accumulators).
// A tile staged with async global->LDS b128 copies when available.
// ---------------------------------------------------------------------------
__global__ __launch_bounds__(256) void gemm_wmma(
    const _Float16* __restrict__ A, const float* __restrict__ W,
    const float* __restrict__ bias, const float* __restrict__ residual,
    float* __restrict__ Cf, _Float16* __restrict__ Ch,
    int M, int N, int K, int gelu) {
  __shared__ _Float16 As[128][48];   // 96B row stride, 16B aligned chunks
  __shared__ _Float16 Bs[128][48];   // W tile stored N-major (transposed)

  const int tid  = threadIdx.x;
  const int lane = tid & 31, wave = tid >> 5;
  const int wm = wave >> 1, wn = wave & 1;
  const int bm = blockIdx.y * 128, bn = blockIdx.x * 128;
  const int hi = lane >> 4, lo = lane & 15;

  // A stage: 512 16-byte chunks (128 rows x 4 chunks); thread does 2 chunks.
  const int ar0 = tid >> 2,          ac0 = (tid & 3) * 8;
  const int ar1 = (tid + 256) >> 2,  ac1 = (tid & 3) * 8;  // (tid+256)&3 == tid&3
  const int nrem = N - bn;

  const _Float16* Abase = A + (size_t)bm * K;
  const float*    Wbase = W + bn;

  v8f acc[2][4] = {};

  for (int k0 = 0; k0 < K; k0 += 32) {
    const _Float16* Ak = Abase + k0;
#if USE_ASYNC_COPY
    async_copy_b128(Ak + ar0 * K + ac0, &As[ar0][ac0]);
    async_copy_b128(Ak + ar1 * K + ac1, &As[ar1][ac1]);
#else
    *(v8h*)&As[ar0][ac0] = *(const v8h*)(Ak + ar0 * K + ac0);
    *(v8h*)&As[ar1][ac1] = *(const v8h*)(Ak + ar1 * K + ac1);
#endif
    const float* Wk = Wbase + (size_t)k0 * N;
    for (int i = 0; i < 16; i++) {
      int idx = tid + i * 256;               // 32x128 tile
      int kk = idx >> 7, n = idx & 127;
      float w = (n < nrem) ? Wk[kk * N + n] : 0.f;
      Bs[n][kk] = (_Float16)w;
    }
    if (k0 + 32 < K) __builtin_prefetch(Wk + 32 * N + (tid & 127), 0, 1);
#if USE_ASYNC_COPY
    wait_async0();
#endif
    __syncthreads();

    v16h af[2], bf[4];
    for (int i = 0; i < 2; i++) {
      const int m = wm * 32 + i * 16 + lo;
      const v8h p0 = *(const v8h*)&As[m][hi * 8];
      const v8h p1 = *(const v8h*)&As[m][hi * 8 + 16];
      for (int e = 0; e < 8; e++) { af[i][e] = p0[e]; af[i][8 + e] = p1[e]; }
    }
    for (int j = 0; j < 4; j++) {
      const int n = wn * 64 + j * 16 + lo;
      const v8h p0 = *(const v8h*)&Bs[n][hi * 8];
      const v8h p1 = *(const v8h*)&Bs[n][hi * 8 + 16];
      for (int e = 0; e < 8; e++) { bf[j][e] = p0[e]; bf[j][8 + e] = p1[e]; }
    }
    for (int i = 0; i < 2; i++)
      for (int j = 0; j < 4; j++)
        acc[i][j] = wmma_f16(af[i], bf[j], acc[i][j]);
    __syncthreads();
  }

  // Epilogue: C layout -> lane lo = column, VGPR r + hi*8 = row within tile.
  for (int i = 0; i < 2; i++)
    for (int j = 0; j < 4; j++) {
      const int n = bn + wn * 64 + j * 16 + lo;
      if (n >= N) continue;
      const float bv = bias ? bias[n] : 0.f;
      for (int r = 0; r < 8; r++) {
        const int m = bm + wm * 32 + i * 16 + hi * 8 + r;
        float vv = acc[i][j][r] + bv;
        if (gelu) vv = 0.5f * vv * (1.f + erff(vv * 0.70710678118f));
        if (Ch) {
          Ch[(size_t)m * N + n] = (_Float16)vv;
        } else {
          if (residual) vv += residual[(size_t)m * N + n];
          Cf[(size_t)m * N + n] = vv;
        }
      }
    }
}

// ---------------------------------------------------------------------------
// Causal attention, flash style, all-f16 I/O. One block = 4 waves = 64 query
// rows of one (b,h); K/V tiles (32 keys x 64 dh) staged cooperatively in LDS
// (K via async copy, V transposed to dh-major). 32 keys per step.
// ---------------------------------------------------------------------------
__global__ __launch_bounds__(128) void attn_k(const _Float16* __restrict__ q,
                                              const _Float16* __restrict__ k,
                                              const _Float16* __restrict__ v,
                                              _Float16* __restrict__ o,
                                              int Hh, int S, int D) {
  __shared__ _Float16 Ks[32][72];      // [key][dh], 144B stride
  __shared__ _Float16 Vs[64][40];      // [dh][key], 80B stride
  __shared__ _Float16 Pl[4][16][48];   // per-wave P transpose staging
  const int tid  = threadIdx.x;
  const int lane = tid & 31, wave = tid >> 5;
  const int hi = lane >> 4, lo = lane & 15;
  const int QB = S / 64;
  const int qtb = blockIdx.x % QB;
  const int tmp = blockIdx.x / QB;
  const int h = tmp % Hh, b = tmp / Hh;
  const int q0b = qtb * 64;
  const int q0  = q0b + wave * 16;
  const float scale = 0.125f;          // 1/sqrt(64)
  const size_t headoff = (size_t)(b * S) * D + h * 64;

  // Q fragments (A layout): lane lo = row, hi selects K-halves of each chunk.
  v16h qa0, qa1;
  {
    const _Float16* qp = q + headoff + (size_t)(q0 + lo) * D;
    const int base = hi * 8;
    const v8h t0 = *(const v8h*)(qp + base);
    const v8h t1 = *(const v8h*)(qp + base + 16);
    const v8h t2 = *(const v8h*)(qp + 32 + base);
    const v8h t3 = *(const v8h*)(qp + 32 + base + 16);
    for (int e = 0; e < 8; e++) {
      qa0[e] = t0[e]; qa0[8 + e] = t1[e];
      qa1[e] = t2[e]; qa1[8 + e] = t3[e];
    }
  }

  v8f oacc[4] = {};
  float mrow[8], lrow[8];
  for (int r = 0; r < 8; r++) { mrow[r] = -1e30f; lrow[r] = 0.f; }

  // Cooperative staging map: thread -> key row, 16-dh slice.
  const int skey = tid >> 2, sdh = (tid & 3) * 16;

  const int nkt = q0b / 32 + 2;        // keys up to q0b+63 (causal max of block)
  for (int kt = 0; kt < nkt; kt++) {
    const int k0 = kt * 32;
    {
      const _Float16* kp = k + headoff + (size_t)(k0 + skey) * D + sdh;
#if USE_ASYNC_COPY
      async_copy_b128(kp,     &Ks[skey][sdh]);
      async_copy_b128(kp + 8, &Ks[skey][sdh + 8]);
#else
      *(v8h*)&Ks[skey][sdh]     = *(const v8h*)kp;
      *(v8h*)&Ks[skey][sdh + 8] = *(const v8h*)(kp + 8);
#endif
      const _Float16* vp = v + headoff + (size_t)(k0 + skey) * D + sdh;
      const v8h b0 = *(const v8h*)vp;
      const v8h b1 = *(const v8h*)(vp + 8);
      for (int e = 0; e < 8; e++) {          // transpose V to dh-major
        Vs[sdh + e][skey]     = b0[e];
        Vs[sdh + 8 + e][skey] = b1[e];
      }
    }
#if USE_ASYNC_COPY
    wait_async0();
#endif
    __syncthreads();

    // Score B-fragments from LDS (lane lo = key column, hi = dh half).
    v16h kb00, kb01, kb10, kb11;
    {
      const v8h x0 = *(const v8h*)&Ks[lo][hi * 8];
      const v8h x1 = *(const v8h*)&Ks[lo][hi * 8 + 16];
      const v8h y0 = *(const v8h*)&Ks[lo][32 + hi * 8];
      const v8h y1 = *(const v8h*)&Ks[lo][32 + hi * 8 + 16];
      const v8h z0 = *(const v8h*)&Ks[16 + lo][hi * 8];
      const v8h z1 = *(const v8h*)&Ks[16 + lo][hi * 8 + 16];
      const v8h w0 = *(const v8h*)&Ks[16 + lo][32 + hi * 8];
      const v8h w1 = *(const v8h*)&Ks[16 + lo][32 + hi * 8 + 16];
      for (int e = 0; e < 8; e++) {
        kb00[e] = x0[e]; kb00[8 + e] = x1[e];
        kb01[e] = y0[e]; kb01[8 + e] = y1[e];
        kb10[e] = z0[e]; kb10[8 + e] = z1[e];
        kb11[e] = w0[e]; kb11[8 + e] = w1[e];
      }
    }
    v8f s0 = {}, s1 = {};
    s0 = wmma_f16(qa0, kb00, s0); s0 = wmma_f16(qa1, kb01, s0);
    s1 = wmma_f16(qa0, kb10, s1); s1 = wmma_f16(qa1, kb11, s1);

    // Online softmax (rows hi*8+r live in this lane's 16-lane half).
    for (int r = 0; r < 8; r++) {
      const int qrow = q0 + hi * 8 + r;
      float a0 = (k0 + lo      <= qrow) ? s0[r] * scale : -1e30f;
      float a1 = (k0 + 16 + lo <= qrow) ? s1[r] * scale : -1e30f;
      float mx = fmaxf(a0, a1);
      for (int msk = 1; msk < 16; msk <<= 1)
        mx = fmaxf(mx, __shfl_xor(mx, msk, 16));
      const float mnew  = fmaxf(mrow[r], mx);
      const float alpha = __expf(mrow[r] - mnew);
      const float p0 = __expf(a0 - mnew);
      const float p1 = __expf(a1 - mnew);
      float ps = p0 + p1;
      for (int msk = 1; msk < 16; msk <<= 1)
        ps += __shfl_xor(ps, msk, 16);
      lrow[r] = lrow[r] * alpha + ps;
      mrow[r] = mnew;
      for (int t = 0; t < 4; t++) oacc[t][r] *= alpha;
      s0[r] = p0; s1[r] = p1;
    }

    // Transpose P (C layout -> A layout) through per-wave LDS tile.
    for (int r = 0; r < 8; r++) {
      Pl[wave][hi * 8 + r][lo]      = (_Float16)s0[r];
      Pl[wave][hi * 8 + r][16 + lo] = (_Float16)s1[r];
    }
    asm volatile("s_wait_dscnt 0x0" ::: "memory");   // CDNA5 split DS counter
    __builtin_amdgcn_wave_barrier();
    v16h pa;
    {
      const v8h p0v = *(const v8h*)&Pl[wave][lo][hi * 8];
      const v8h p1v = *(const v8h*)&Pl[wave][lo][hi * 8 + 16];
      for (int e = 0; e < 8; e++) { pa[e] = p0v[e]; pa[8 + e] = p1v[e]; }
    }
    __builtin_amdgcn_wave_barrier();

    // O += P @ V  (V B-fragments contiguous in dh-major LDS tile).
    for (int t = 0; t < 4; t++) {
      const v8h w0 = *(const v8h*)&Vs[t * 16 + lo][hi * 8];
      const v8h w1 = *(const v8h*)&Vs[t * 16 + lo][hi * 8 + 16];
      v16h vbf;
      for (int e = 0; e < 8; e++) { vbf[e] = w0[e]; vbf[8 + e] = w1[e]; }
      oacc[t] = wmma_f16(pa, vbf, oacc[t]);
    }
    __syncthreads();   // K/V tiles consumed; safe to restage next iteration
  }

  // Normalize and store (f16, C layout).
  for (int t = 0; t < 4; t++)
    for (int r = 0; r < 8; r++) {
      const int m = q0 + hi * 8 + r;
      o[headoff + (size_t)m * D + t * 16 + lo] = (_Float16)(oacc[t][r] / lrow[r]);
    }
}

// ---------------------------------------------------------------------------
// Host orchestration.
// ---------------------------------------------------------------------------
extern "C" void kernel_launch(void* const* d_in, const int* in_sizes, int n_in,
                              void* d_out, int out_size, void* d_ws, size_t ws_size,
                              hipStream_t stream) {
  (void)in_sizes; (void)n_in; (void)out_size; (void)ws_size;
  const int*   tokens  = (const int*)  d_in[0];
  const float* tok_emb = (const float*)d_in[1];
  const float* pos_emb = (const float*)d_in[2];
  const float* ln1_g   = (const float*)d_in[3];
  const float* ln1_b   = (const float*)d_in[4];
  const float* wq      = (const float*)d_in[5];
  const float* bq      = (const float*)d_in[6];
  const float* wk      = (const float*)d_in[7];
  const float* bk      = (const float*)d_in[8];
  const float* wv      = (const float*)d_in[9];
  const float* bv      = (const float*)d_in[10];
  const float* wo      = (const float*)d_in[11];
  const float* bo      = (const float*)d_in[12];
  const float* ln2_g   = (const float*)d_in[13];
  const float* ln2_b   = (const float*)d_in[14];
  const float* w1      = (const float*)d_in[15];
  const float* b1      = (const float*)d_in[16];
  const float* w2      = (const float*)d_in[17];
  const float* b2      = (const float*)d_in[18];
  const float* fn_g    = (const float*)d_in[19];
  const float* fn_b    = (const float*)d_in[20];
  const float* head_w  = (const float*)d_in[21];
  const float* head_b  = (const float*)d_in[22];
  float* out = (float*)d_out;

  const int Bb = 2, S = 1024, L = 12, Hh = 12, D = 768, F4 = 3072, V = 50257;
  const int M = Bb * S;  // 2048

  char* p = (char*)d_ws;
  float* x  = (float*)p;  p += (size_t)M * D * 4;
  float* x2 = (float*)p;  p += (size_t)M * D * 4;
  _Float16* hbuf = (_Float16*)p; p += (size_t)M * D * 2;
  _Float16* qb   = (_Float16*)p; p += (size_t)M * D * 2;
  _Float16* kb   = (_Float16*)p; p += (size_t)M * D * 2;
  _Float16* vb   = (_Float16*)p; p += (size_t)M * D * 2;
  _Float16* ob   = (_Float16*)p; p += (size_t)M * D * 2;
  _Float16* ff   = (_Float16*)p; p += (size_t)M * F4 * 2;

  embed_k<<<256, 256, 0, stream>>>(tokens, tok_emb, pos_emb, x, M * D, D, S);

  const dim3 gD (D  / 128, M / 128);   // N=768
  const dim3 gF (F4 / 128, M / 128);   // N=3072
  const dim3 gV ((V + 127) / 128, M / 128);
  const int attn_blocks = Bb * Hh * (S / 64);   // 384

  for (int l = 0; l < L; l++) {
    layernorm_k<<<M, 256, 0, stream>>>(x, ln1_g + l * D, ln1_b + l * D, hbuf, D);
    gemm_wmma<<<gD, 256, 0, stream>>>(hbuf, wq + (size_t)l * D * D, bq + l * D,
                                      nullptr, nullptr, qb, M, D, D, 0);
    gemm_wmma<<<gD, 256, 0, stream>>>(hbuf, wk + (size_t)l * D * D, bk + l * D,
                                      nullptr, nullptr, kb, M, D, D, 0);
    gemm_wmma<<<gD, 256, 0, stream>>>(hbuf, wv + (size_t)l * D * D, bv + l * D,
                                      nullptr, nullptr, vb, M, D, D, 0);
    attn_k<<<attn_blocks, 128, 0, stream>>>(qb, kb, vb, ob, Hh, S, D);
    gemm_wmma<<<gD, 256, 0, stream>>>(ob, wo + (size_t)l * D * D, bo + l * D,
                                      x, x2, nullptr, M, D, D, 0);
    layernorm_k<<<M, 256, 0, stream>>>(x2, ln2_g + l * D, ln2_b + l * D, hbuf, D);
    gemm_wmma<<<gF, 256, 0, stream>>>(hbuf, w1 + (size_t)l * D * F4, b1 + l * F4,
                                      nullptr, nullptr, ff, M, F4, D, 1);
    gemm_wmma<<<gD, 256, 0, stream>>>(ff, w2 + (size_t)l * F4 * D, b2 + l * D,
                                      x2, x, nullptr, M, D, F4, 0);
  }

  layernorm_k<<<M, 256, 0, stream>>>(x, fn_g, fn_b, hbuf, D);
  gemm_wmma<<<gV, 256, 0, stream>>>(hbuf, head_w, head_b,
                                    nullptr, out, nullptr, M, V, D, 0);
}
